// GaussianBasis_9268539425092
// MI455X (gfx1250) — compile-verified, matrix operand order checked by reading, not run on hardware
//
#include <hip/hip_runtime.h>
#include <hip/hip_fp16.h>

#define HH 256
#define WW 256
#define NG 10000
#define NPAD 10080          // 5 * 2016, multiple of 32
#define PER_SLICE 2016
#define NSLICE 5

typedef __attribute__((ext_vector_type(2)))  float    v2f;
typedef __attribute__((ext_vector_type(8)))  float    v8f;
typedef __attribute__((ext_vector_type(16))) _Float16 v16h;
typedef __attribute__((ext_vector_type(2)))  __fp16   pk2;   // matches cvt_pkrtz return type

union AlphaA { v16h v; pk2 h[8]; };
union ColB   { v16h v; uint4 q[2]; };

// ---------------------------------------------------------------- init: out = shift
__global__ void gb_init_out(float* __restrict__ out, const float* __restrict__ shift, int n) {
    int i = blockIdx.x * blockDim.x + threadIdx.x;
    float s = shift[0];
    if (i < n) out[i] = s;
}

// ---------------------------------------------------------------- per-gaussian preprocess
// P[n] = {conic_a, conic_b, conic_c, cx, cy, 0, 0, 0} (2 x float4)
// Op[n] = opacity (0 in pad region -> exact tail handling)
// CT[c][n] = half(color[n][c] * scale), row 3 = zeros (B-matrix padding rows)
__global__ void gb_preprocess(const float* __restrict__ xyz, const float* __restrict__ chol,
                              const float* __restrict__ cols, const float* __restrict__ opac,
                              const float* __restrict__ scale,
                              float* __restrict__ P, float* __restrict__ Op,
                              _Float16* __restrict__ CT) {
    int n = blockIdx.x * blockDim.x + threadIdx.x;
    if (n >= NPAD) return;
    float sc = scale[0];
    float4* P4 = (float4*)P;
    if (n < NG) {
        float mx = tanhf(xyz[2 * n]), my = tanhf(xyz[2 * n + 1]);
        float cx = 0.5f * WW * (mx + 1.0f);
        float cy = 0.5f * HH * (my + 1.0f);
        float l0 = chol[3 * n] + 0.5f;
        float l1 = chol[3 * n + 1];
        float l2 = chol[3 * n + 2] + 0.5f;
        float cxx = l0 * l0, cxy = l0 * l1, cyy = l1 * l1 + l2 * l2;
        float det = cxx * cyy - cxy * cxy;
        float inv = 1.0f / det;
        P4[2 * n]     = make_float4(cyy * inv, -cxy * inv, cxx * inv, cx);
        P4[2 * n + 1] = make_float4(cy, 0.f, 0.f, 0.f);
        Op[n] = opac[n];
        CT[0 * NPAD + n] = (_Float16)(cols[3 * n]     * sc);
        CT[1 * NPAD + n] = (_Float16)(cols[3 * n + 1] * sc);
        CT[2 * NPAD + n] = (_Float16)(cols[3 * n + 2] * sc);
        CT[3 * NPAD + n] = (_Float16)0.f;
    } else {
        P4[2 * n]     = make_float4(0.f, 0.f, 0.f, 0.f);
        P4[2 * n + 1] = make_float4(0.f, 0.f, 0.f, 0.f);
        Op[n] = 0.f;   // pad gaussians contribute exactly zero
        CT[0 * NPAD + n] = (_Float16)0.f;
        CT[1 * NPAD + n] = (_Float16)0.f;
        CT[2 * NPAD + n] = (_Float16)0.f;
        CT[3 * NPAD + n] = (_Float16)0.f;
    }
}

__device__ __forceinline__ float gb_gate(float sig, float op) {
    float e  = __builtin_amdgcn_exp2f(sig * -1.4426950408889634f); // exp(-sig)
    float al = op * e;
    float cl = fminf(al, 0.999f);
    return (sig >= 0.f && al >= (1.0f / 255.0f)) ? cl : 0.f;
}

// ---------------------------------------------------------------- main render
// One wave per 16-pixel horizontal strip; 8 waves/block; blockIdx.y = gaussian slice.
__global__ __launch_bounds__(256) void gb_render(const float* __restrict__ P,
                                                 const float* __restrict__ Op,
                                                 const _Float16* __restrict__ CT,
                                                 float* __restrict__ out) {
    const int lane = threadIdx.x & 31;
    const int wave = threadIdx.x >> 5;
    const int c    = lane & 15;
    const bool hi  = lane >= 16;

    const int S  = blockIdx.x * 8 + wave;   // strip id 0..4095
    const int y  = S >> 4;
    const int x0 = (S & 15) << 4;
    const float yf = (float)y, x0f = (float)x0;

    // Sigma B-matrix (4x16 f32, K x pixel): rows {px^2, px, 1, 0} in strip-local coords.
    v2f Bm;
    Bm.x = hi ? 1.0f : (float)(c * c);   // lanes 0-15: K0 ; lanes 16-31: K2
    Bm.y = hi ? 0.0f : (float)c;         // lanes 0-15: K1 ; lanes 16-31: K3

    const int gStart = blockIdx.y * PER_SLICE;
    const int gEnd   = gStart + PER_SLICE;

    v8f acc = {};
    const v8f zc = {};

    const float4* P4 = (const float4*)P;
    const _Float16* ctRow = CT + (size_t)((c < 3) ? c : 3) * NPAD;

    for (int base = gStart; base < gEnd; base += 32) {
        // ---- per-gaussian features for two 16-gaussian tiles (strip-local coords)
        v2f A0, A1;
        {
            int g = base + c;
            float4 pa = P4[2 * g];
            float4 pb = P4[2 * g + 1];
            float a = pa.x, b = pa.y, cc2 = pa.z;
            float cxl = pa.w - x0f, cyl = pb.x - yf;
            float f3 = -(a * cxl + b * cyl);
            float f5 = 0.5f * a * cxl * cxl + 0.5f * cc2 * cyl * cyl + b * cxl * cyl;
            A0.x = hi ? f5  : 0.5f * a;
            A0.y = hi ? 0.f : f3;
        }
        {
            int g = base + 16 + c;
            float4 pa = P4[2 * g];
            float4 pb = P4[2 * g + 1];
            float a = pa.x, b = pa.y, cc2 = pa.z;
            float cxl = pa.w - x0f, cyl = pb.x - yf;
            float f3 = -(a * cxl + b * cyl);
            float f5 = 0.5f * a * cxl * cxl + 0.5f * cc2 * cyl * cyl + b * cxl * cyl;
            A1.x = hi ? f5  : 0.5f * a;
            A1.y = hi ? 0.f : f3;
        }
        if (base + 32 < gEnd) __builtin_prefetch(&P4[2 * (base + 32 + c)], 0, 3);

        // ---- sigma GEMM: D[g, px] = features x monomials  (full f32 matrix core)
        v8f s0 = __builtin_amdgcn_wmma_f32_16x16x4_f32(false, A0, false, Bm, (short)0, zc, false, false);
        v8f s1 = __builtin_amdgcn_wmma_f32_16x16x4_f32(false, A1, false, Bm, (short)0, zc, false, false);

        // ---- color B-matrix (32g x 16col f16): lane = color channel, K packed in regs
        ColB cb;
        const _Float16* cp = ctRow + base + (hi ? 8 : 0);
        cb.q[0] = *(const uint4*)(cp);        // K = {0..7}  (+8 for hi half)
        cb.q[1] = *(const uint4*)(cp + 16);   // K = {16..23}(+8 for hi half)

        // ---- elementwise exp/gate, pack f32->f16; sigma-D layout == alpha-A layout
        AlphaA aa;
        #pragma unroll
        for (int j = 0; j < 4; ++j) {
            float oL0 = Op[base + 2 * j],          oH0 = Op[base + 2 * j + 8];
            float oL1 = Op[base + 2 * j + 1],      oH1 = Op[base + 2 * j + 9];
            float a0 = gb_gate(s0[2 * j],     hi ? oH0 : oL0);
            float a1 = gb_gate(s0[2 * j + 1], hi ? oH1 : oL1);
            aa.h[j] = __builtin_amdgcn_cvt_pkrtz(a0, a1);

            float pL0 = Op[base + 16 + 2 * j],     pH0 = Op[base + 16 + 2 * j + 8];
            float pL1 = Op[base + 16 + 2 * j + 1], pH1 = Op[base + 16 + 2 * j + 9];
            float b0 = gb_gate(s1[2 * j],     hi ? pH0 : pL0);
            float b1 = gb_gate(s1[2 * j + 1], hi ? pH1 : pL1);
            aa.h[4 + j] = __builtin_amdgcn_cvt_pkrtz(b0, b1);
        }

        // ---- color accumulation GEMM: acc[px, col] += alpha x colors
        acc = __builtin_amdgcn_wmma_f32_16x16x32_f16(false, aa.v, false, cb.v, (short)0, acc, false, false);
    }

    // ---- merge into image (scale folded into colors, shift applied by init kernel)
    if (c < 3) {
        float* op = out + (size_t)c * (HH * WW) + (size_t)y * WW + x0;
        #pragma unroll
        for (int r = 0; r < 8; ++r) {
            int p = r + (hi ? 8 : 0);
            atomicAdd(op + p, acc[r]);
        }
    }
}

// ---------------------------------------------------------------- launcher
extern "C" void kernel_launch(void* const* d_in, const int* in_sizes, int n_in,
                              void* d_out, int out_size, void* d_ws, size_t ws_size,
                              hipStream_t stream) {
    (void)in_sizes; (void)n_in; (void)out_size; (void)ws_size;
    const float* xyz   = (const float*)d_in[0];
    const float* chol  = (const float*)d_in[1];
    const float* cols  = (const float*)d_in[2];
    const float* opac  = (const float*)d_in[3];
    const float* scale = (const float*)d_in[4];
    const float* shift = (const float*)d_in[5];
    float* out = (float*)d_out;

    char* ws = (char*)d_ws;
    float*    P  = (float*)(ws);                     // NPAD*8 f32  = 322560 B (256-aligned)
    float*    Op = (float*)(ws + 322560);            // NPAD   f32  =  40320 B
    _Float16* CT = (_Float16*)(ws + 322560 + 40448); // 4*NPAD f16  =  80640 B

    gb_init_out<<<(3 * HH * WW + 255) / 256, 256, 0, stream>>>(out, shift, 3 * HH * WW);
    gb_preprocess<<<(NPAD + 255) / 256, 256, 0, stream>>>(xyz, chol, cols, opac, scale, P, Op, CT);

    dim3 grid(512, NSLICE, 1);   // 512 blocks * 8 waves = 4096 strips of 16 pixels
    gb_render<<<grid, 256, 0, stream>>>(P, Op, CT, out);
}
